// Muti_Stacked_Attention_48430051230377
// MI455X (gfx1250) — matrix operand order repaired and checked by reading, not compiled, not run on hardware
//
#include <hip/hip_runtime.h>

typedef __bf16 bf16;
typedef __attribute__((ext_vector_type(16))) __bf16 v16bf;
typedef __attribute__((ext_vector_type(8)))  __bf16 v8bf;
typedef __attribute__((ext_vector_type(8)))  float  v8f;

#define NROWS 65536
#define EPSV 1e-5f

// ---------------- conversion / init kernels ----------------
__global__ void k_zero_bf16(bf16* p, size_t n) {
  size_t i = (size_t)blockIdx.x * 256 + threadIdx.x;
  if (i < n) p[i] = (bf16)0.f;
}

__global__ void k_f32_to_bf16(const float* __restrict__ s, bf16* __restrict__ d, size_t n) {
  size_t i = (size_t)blockIdx.x * 256 + threadIdx.x;
  if (i < n) d[i] = (bf16)s[i];
}

// weight (cout, K) fp32 -> (cout, Kp) bf16, zero padded K..Kp
__global__ void k_weight_to_bf16_pad(const float* __restrict__ w, bf16* __restrict__ d,
                                     int K, int Kp, size_t n /* cout*Kp */) {
  size_t i = (size_t)blockIdx.x * 256 + threadIdx.x;
  if (i >= n) return;
  int kk = (int)(i % (size_t)Kp);
  size_t r = i / (size_t)Kp;
  d[i] = (kk < K) ? (bf16)w[r * (size_t)K + kk] : (bf16)0.f;
}

// src [rows, C] fp32 (contiguous) -> dst bf16 at dst[row*ld + colOff + c]
__global__ void k_f32_to_bf16_strided(const float* __restrict__ s, int C,
                                      bf16* __restrict__ d, int ld, int colOff, size_t n) {
  size_t i = (size_t)blockIdx.x * 256 + threadIdx.x;
  if (i >= n) return;
  int c = (int)(i % (size_t)C);
  size_t r = i / (size_t)C;
  d[r * (size_t)ld + colOff + c] = (bf16)s[i];
}

// ---------------- WMMA GEMM + BN partial sums ----------------
// Y[M=65536, Ncols] = A[M,K](bf16,lda) * W^T + bias ; W stored (Ncols, K) row-major bf16 (ldb=Kp)
// Per-wave 16 x (16*CT) tile, 8 waves/block (128 rows). B fragments are loaded in
// groups of up to 4 distinct register sets so loads overlap and WMMA waits are partial.
// Also writes deterministic per-column partial sum/sumsq at unique (pidx,col) slots.
template<int CT>
__global__ __launch_bounds__(256) void gemm_bn(
    const bf16* __restrict__ A, int lda,
    const bf16* __restrict__ B, int ldb,
    const float* __restrict__ bias,
    float* __restrict__ Y,
    float* __restrict__ psum, float* __restrict__ psq,
    int K, int Ncols) {
  constexpr int GR = (CT < 4) ? CT : 4;   // B-fragment group size
  const int wave = threadIdx.x >> 5;
  const int lane = threadIdx.x & 31;
  const int lr   = lane & 15;
  const int half = lane >> 4;
  const int row0 = (blockIdx.y * 8 + wave) * 16;
  const int col0 = blockIdx.x * (16 * CT);

  v8f acc[CT];
#pragma unroll
  for (int t = 0; t < CT; ++t) acc[t] = {};

  const bf16* Arow = A + (size_t)(row0 + lr) * lda + half * 8;
  const bf16* Bcol = B + (size_t)(col0 + lr) * ldb + half * 8;

  for (int k = 0; k < K; k += 32) {
    union { v16bf v; v8bf h[2]; } a;
    a.h[0] = *(const v8bf*)(Arow + k);
    a.h[1] = *(const v8bf*)(Arow + k + 16);
#pragma unroll
    for (int g = 0; g < CT / GR; ++g) {
      union { v16bf v; v8bf h[2]; } b[GR];
#pragma unroll
      for (int t = 0; t < GR; ++t) {
        const bf16* Bp = Bcol + (size_t)(g * GR + t) * 16 * ldb + k;
        b[t].h[0] = *(const v8bf*)(Bp);
        b[t].h[1] = *(const v8bf*)(Bp + 16);
      }
#pragma unroll
      for (int t = 0; t < GR; ++t)
        acc[g * GR + t] = __builtin_amdgcn_wmma_f32_16x16x32_bf16(
            false, a.v, false, b[t].v, (short)0, acc[g * GR + t], false, false);
    }
  }

  const int pidx = (blockIdx.y * 8 + wave) * 2 + half;  // 8 rows per pidx
#pragma unroll
  for (int t = 0; t < CT; ++t) {
    int col = col0 + t * 16 + lr;
    float bv = bias ? bias[col] : 0.f;
    float s = 0.f, sq = 0.f;
#pragma unroll
    for (int r = 0; r < 8; ++r) {
      int row = row0 + r + half * 8;
      float v = acc[t][r] + bv;
      Y[(size_t)row * Ncols + col] = v;
      s += v; sq += v * v;
    }
    psum[(size_t)pidx * Ncols + col] = s;
    psq [(size_t)pidx * Ncols + col] = sq;
  }
}

// ---------------- BN finalize: one block per column ----------------
__global__ void bn_finalize(const float* __restrict__ psum, const float* __restrict__ psq,
                            int P, int Ncols, float* __restrict__ mu, float* __restrict__ rstd,
                            float inv_n) {
  __shared__ float ss[256], sk[256];
  int c = blockIdx.x;
  float s = 0.f, q = 0.f;
  for (int p = threadIdx.x; p < P; p += 256) {
    s += psum[(size_t)p * Ncols + c];
    q += psq [(size_t)p * Ncols + c];
  }
  ss[threadIdx.x] = s; sk[threadIdx.x] = q;
  __syncthreads();
  for (int st = 128; st > 0; st >>= 1) {
    if ((int)threadIdx.x < st) { ss[threadIdx.x] += ss[threadIdx.x + st]; sk[threadIdx.x] += sk[threadIdx.x + st]; }
    __syncthreads();
  }
  if (threadIdx.x == 0) {
    float m = ss[0] * inv_n;
    float v = sk[0] * inv_n - m * m;
    mu[c] = m;
    rstd[c] = rsqrtf(v + EPSV);
  }
}

// ---------------- normalize + ReLU, optional bf16 and fp32 outputs ----------------
__global__ void bn_relu_apply(const float* __restrict__ Y, int Ncols,
                              const float* __restrict__ g, const float* __restrict__ be,
                              const float* __restrict__ mu, const float* __restrict__ rstd,
                              bf16* dstB, int ldB, int colOffB,
                              float* dstF, int ldF, size_t total) {
  size_t i = (size_t)blockIdx.x * 256 + threadIdx.x;
  if (i >= total) return;
  int col = (int)(i % (size_t)Ncols);
  size_t row = i / (size_t)Ncols;
  float y = Y[i];
  float o = g[col] * (y - mu[col]) * rstd[col] + be[col];
  o = fmaxf(o, 0.f);
  if (dstB) dstB[row * (size_t)ldB + colOffB + col] = (bf16)o;
  if (dstF) dstF[row * (size_t)ldF + col] = o;
}

// ---------------- 4-head attention, one wave per row ----------------
__global__ __launch_bounds__(256) void attn4(
    const float* __restrict__ q,  // [N,128] fp32, cols 0..dh valid
    const float* __restrict__ k,  // [N,4*dh] fp32
    const float* __restrict__ v,  // [N,4*dh] fp32 (feats)
    bf16* __restrict__ pf, int colOff, int dh, int nrows) {
  int wave = threadIdx.x >> 5;
  int lane = threadIdx.x & 31;
  int row = blockIdx.x * 8 + wave;
  if (row >= nrows) return;
  const float* qr = q + (size_t)row * 128;
  const float* kr = k + (size_t)row * (4 * dh);
  const float* vr = v + (size_t)row * (4 * dh);
  float d0 = 0.f, d1 = 0.f, d2 = 0.f, d3 = 0.f;
  for (int x = lane; x < dh; x += 32) {
    float qv = qr[x];
    d0 += qv * kr[x];
    d1 += qv * kr[dh + x];
    d2 += qv * kr[2 * dh + x];
    d3 += qv * kr[3 * dh + x];
  }
#pragma unroll
  for (int off = 16; off > 0; off >>= 1) {
    d0 += __shfl_xor(d0, off, 32);
    d1 += __shfl_xor(d1, off, 32);
    d2 += __shfl_xor(d2, off, 32);
    d3 += __shfl_xor(d3, off, 32);
  }
  float sc = rsqrtf((float)dh);
  float l0 = d0 * sc, l1 = d1 * sc, l2 = d2 * sc, l3 = d3 * sc;
  float m = fmaxf(fmaxf(l0, l1), fmaxf(l2, l3));
  float e0 = expf(l0 - m), e1 = expf(l1 - m), e2 = expf(l2 - m), e3 = expf(l3 - m);
  float inv = 1.f / (e0 + e1 + e2 + e3);
  float w0 = e0 * inv, w1 = e1 * inv, w2 = e2 * inv, w3 = e3 * inv;
  bf16* prow = pf + (size_t)row * 1024 + colOff;
  for (int x = lane; x < dh; x += 32) {
    float o = w0 * vr[x] + w1 * vr[dh + x] + w2 * vr[2 * dh + x] + w3 * vr[3 * dh + x];
    bf16 ob = (bf16)o;
    prow[x] = ob;
    prow[dh + x] = ob;
    prow[2 * dh + x] = ob;
    prow[3 * dh + x] = ob;
  }
}

// ---------------- host-side dispatch helpers ----------------
static void run_gemm(const bf16* A, int lda, const bf16* B, int ldb, const float* bias,
                     float* Y, float* psum, float* psq, int K, int Ncols, hipStream_t s) {
  dim3 blk(256, 1, 1);
  dim3 grd(1, NROWS / 128, 1);
  if (Ncols % 128 == 0) {
    grd.x = Ncols / 128;
    gemm_bn<8><<<grd, blk, 0, s>>>(A, lda, B, ldb, bias, Y, psum, psq, K, Ncols);
  } else if (Ncols % 64 == 0) {
    grd.x = Ncols / 64;
    gemm_bn<4><<<grd, blk, 0, s>>>(A, lda, B, ldb, bias, Y, psum, psq, K, Ncols);
  } else if (Ncols % 32 == 0) {
    grd.x = Ncols / 32;
    gemm_bn<2><<<grd, blk, 0, s>>>(A, lda, B, ldb, bias, Y, psum, psq, K, Ncols);
  } else {
    grd.x = Ncols / 16;
    gemm_bn<1><<<grd, blk, 0, s>>>(A, lda, B, ldb, bias, Y, psum, psq, K, Ncols);
  }
}

static void run_bn(const float* Y, int Ncols, const float* g, const float* be,
                   float* psum, float* psq, float* mu, float* rstd,
                   bf16* dstB, int ldB, int colOffB, float* dstF, int ldF, hipStream_t s) {
  bn_finalize<<<Ncols, 256, 0, s>>>(psum, psq, NROWS / 8, Ncols, mu, rstd, 1.f / NROWS);
  size_t total = (size_t)NROWS * Ncols;
  bn_relu_apply<<<(unsigned)((total + 255) / 256), 256, 0, s>>>(
      Y, Ncols, g, be, mu, rstd, dstB, ldB, colOffB, dstF, ldF, total);
}

static inline unsigned gblk(size_t n) { return (unsigned)((n + 255) / 256); }

extern "C" void kernel_launch(void* const* d_in, const int* in_sizes, int n_in,
                              void* d_out, int out_size, void* d_ws, size_t ws_size,
                              hipStream_t stream) {
  (void)in_sizes; (void)n_in; (void)out_size; (void)ws_size;
  const int CHv[5] = {64, 64, 128, 256, 512};
  const int pfOff[5] = {0, 64, 128, 256, 512};
  const int N = NROWS;

  const float* f[5];
  for (int j = 0; j < 5; ++j) f[j] = (const float*)d_in[j];
  const float *dq_w[4], *dq_b[4], *dq_g[4], *dq_be[4];
  const float *sq_w[4], *sq_b[4], *sq_g[4], *sq_be[4];
  for (int j = 0; j < 4; ++j) {
    int b = 5 + 8 * j;
    dq_w[j] = (const float*)d_in[b + 0]; dq_b[j]  = (const float*)d_in[b + 1];
    dq_g[j] = (const float*)d_in[b + 2]; dq_be[j] = (const float*)d_in[b + 3];
    sq_w[j] = (const float*)d_in[b + 4]; sq_b[j]  = (const float*)d_in[b + 5];
    sq_g[j] = (const float*)d_in[b + 6]; sq_be[j] = (const float*)d_in[b + 7];
  }
  const float *dk_w[4], *dk_b[4], *dk_g[4], *dk_be[4];
  for (int t = 0; t < 4; ++t) {
    int b = 37 + 4 * t;
    dk_w[t] = (const float*)d_in[b + 0]; dk_b[t]  = (const float*)d_in[b + 1];
    dk_g[t] = (const float*)d_in[b + 2]; dk_be[t] = (const float*)d_in[b + 3];
  }
  const float* fus_w = (const float*)d_in[53];
  const float* fus_g = (const float*)d_in[54];
  const float* fus_b = (const float*)d_in[55];

  // ---- workspace carve ----
  char* base = (char*)d_ws;
  size_t off = 0;
  auto carve = [&](size_t bytes) -> char* {
    char* p = base + off;
    off += (bytes + 255) & ~(size_t)255;
    return p;
  };
  bf16* pf = (bf16*)carve((size_t)N * 1024 * 2);       // concat activations (bf16)
  bf16* fb[5]; fb[0] = nullptr;
  for (int i = 1; i < 5; ++i) fb[i] = (bf16*)carve((size_t)N * CHv[i] * 2);
  bf16*  inp = (bf16*)carve((size_t)N * 128 * 2);      // [h | prev_q], ld=128
  float* qf  = (float*)carve((size_t)N * 128 * 4);
  float* kf  = (float*)carve((size_t)N * 512 * 4);
  float* yf  = (float*)carve((size_t)N * 512 * 4);     // pre-BN GEMM output (stages)
  const int P = N / 8;
  float* psum = (float*)carve((size_t)P * 1024 * 4);
  float* psq  = (float*)carve((size_t)P * 1024 * 4);
  float* mu   = (float*)carve(1024 * 4);
  float* rstd = (float*)carve(1024 * 4);
  bf16* dqW[4]; bf16* sqW[4]; int sqKp[4]; bf16* dkW[4];
  for (int j = 0; j < 4; ++j) dqW[j] = (bf16*)carve((size_t)(CHv[j] / 4) * CHv[j] * 2);
  for (int j = 0; j < 4; ++j) {
    int dsq = CHv[j + 1] / 4;
    sqKp[j] = dsq < 32 ? 32 : dsq;
    sqW[j] = (bf16*)carve((size_t)dsq * sqKp[j] * 2);
  }
  for (int t = 0; t < 4; ++t) dkW[t] = (bf16*)carve((size_t)CHv[t + 1] * CHv[t + 1] * 2);
  bf16* fusW = (bf16*)carve((size_t)1024 * 1024 * 2);

  // ---- prep: zero pad region, convert inputs & weights to bf16 ----
  k_zero_bf16<<<gblk((size_t)N * 128), 256, 0, stream>>>(inp, (size_t)N * 128);
  k_f32_to_bf16_strided<<<gblk((size_t)N * 64), 256, 0, stream>>>(f[0], 64, pf, 1024, 0, (size_t)N * 64);
  for (int i = 1; i < 5; ++i)
    k_f32_to_bf16<<<gblk((size_t)N * CHv[i]), 256, 0, stream>>>(f[i], fb[i], (size_t)N * CHv[i]);
  for (int j = 0; j < 4; ++j) {
    int co = CHv[j] / 4, K = CHv[j];
    k_weight_to_bf16_pad<<<gblk((size_t)co * K), 256, 0, stream>>>(dq_w[j], dqW[j], K, K, (size_t)co * K);
    int dsq = CHv[j + 1] / 4;
    k_weight_to_bf16_pad<<<gblk((size_t)dsq * sqKp[j]), 256, 0, stream>>>(sq_w[j], sqW[j], dsq, sqKp[j], (size_t)dsq * sqKp[j]);
    int c = CHv[j + 1];
    k_weight_to_bf16_pad<<<gblk((size_t)c * c), 256, 0, stream>>>(dk_w[j], dkW[j], c, c, (size_t)c * c);
  }
  k_weight_to_bf16_pad<<<gblk((size_t)1024 * 1024), 256, 0, stream>>>(fus_w, fusW, 1024, 1024, (size_t)1024 * 1024);

  // ---- four attention stages ----
  for (int i = 1; i <= 4; ++i) {
    int j = i - 1;
    int cprev = CHv[j], c = CHv[i];
    int wh = cprev / 4, dh = c / 4;

    // dq: res[i-1] (pf slice) -> h  (bf16 into inp[:, 0:wh])
    run_gemm(pf + pfOff[j], 1024, dqW[j], cprev, dq_b[j], yf, psum, psq, cprev, wh, stream);
    run_bn(yf, wh, dq_g[j], dq_be[j], psum, psq, mu, rstd, inp, 128, 0, nullptr, 0, stream);

    // sq: [h | prev_q] -> q  (fp32 into qf; bf16 into inp[:, dh:2dh] for next stage)
    int Ksq = (i == 1) ? 32 : dh;  // padded K for the 16x16 stage-1 weight
    run_gemm(inp, 128, sqW[j], Ksq, sq_b[j], yf, psum, psq, Ksq, dh, stream);
    run_bn(yf, dh, sq_g[j], sq_be[j], psum, psq, mu, rstd,
           (i < 4) ? inp : nullptr, 128, dh, qf, 128, stream);

    // dk: feats[i] -> k  (fp32 into kf)
    run_gemm(fb[i], c, dkW[j], c, dk_b[j], yf, psum, psq, c, c, stream);
    run_bn(yf, c, dk_g[j], dk_be[j], psum, psq, mu, rstd, nullptr, 0, 0, kf, c, stream);

    // attention: softmax over 4 heads, write tiled result (bf16) into pf slice
    attn4<<<N / 8, 256, 0, stream>>>(qf, kf, f[i], pf, pfOff[i], dh, N);
  }

  // ---- fusion: pf @ fus_w.T (no bias), BN+ReLU in place on d_out ----
  float* out = (float*)d_out;
  run_gemm(pf, 1024, fusW, 1024, nullptr, out, psum, psq, 1024, 1024, stream);
  run_bn(out, 1024, fus_g, fus_b, psum, psq, mu, rstd, nullptr, 0, 0, out, 1024, stream);
}
